// BiLSTM_CRF_31026843746485
// MI455X (gfx1250) — matrix operand order repaired and checked by reading, not compiled, not run on hardware
//
#include <hip/hip_runtime.h>
#include <hip/hip_bf16.h>

typedef __attribute__((ext_vector_type(16))) __bf16 v16bf;
typedef __attribute__((ext_vector_type(8)))  float  v8f;

#define BB   64
#define SS   512
#define WW   16
#define EE   300
#define TE   50
#define CE   50
#define CC   50
#define HID  256
#define TGT  16
#define KIN  400
#define KPAD 416
#define G4   1024
#define HPAD 264   // 256 + 8 bf16 pad -> row stride 528B, conflict-free column reads

// ---------------- fragment loaders (CDNA5 wave32 WMMA layouts) ----------------

// A fragment (16x32 bf16, row-major source [M][ld]): lane r holds row row0+r.
// element e -> VGPR p=e>>1 ; K = (p<4 ? 2p : 16+2(p-4)) + hi*8 + (e&1)
__device__ __forceinline__ v16bf load_a_frag(const __bf16* base, int ld,
                                             int row0, int k0, int r, int hi) {
  v16bf a;
  const __bf16* p = base + (size_t)(row0 + r) * ld + k0 + hi * 8;
#pragma unroll
  for (int e = 0; e < 16; ++e) {
    int pr = e >> 1;
    int k = ((pr < 4) ? (pr * 2) : (16 + (pr - 4) * 2)) + (e & 1);
    a[e] = p[k];
  }
  return a;
}

// B fragment (32x16 bf16) where B(k,n) = W[n][k], W row-major [N][ld]:
// lane r holds column n0+r ; K = k0 + hi*8 + (e<8 ? e : e+8)  (same k order as A)
__device__ __forceinline__ v16bf load_b_fragT(const __bf16* base, int ld,
                                              int n0, int k0, int r, int hi) {
  v16bf b;
  const __bf16* p = base + (size_t)(n0 + r) * ld + k0 + hi * 8;
#pragma unroll
  for (int e = 0; e < 16; ++e) b[e] = p[(e < 8) ? e : (e + 8)];
  return b;
}

__device__ __forceinline__ float sigf(float x) { return 1.0f / (1.0f + __expf(-x)); }

// gpre fragment-tiled layout: [dir][row-tile 0..2047][n-tile 0..63][lane 0..31][8]
__device__ __forceinline__ size_t gpre_base(int dir, int mtile, int ntile) {
  return (((size_t)dir * 2048 + mtile) * 64 + ntile) * 256;
}

// ---------------- 1) features: embeddings + char conv -> bf16 feat [S*B][416] --
__global__ void featurize_kernel(const int* __restrict__ X, const int* __restrict__ tags,
                                 const int* __restrict__ chars,
                                 const float* __restrict__ embed_table,
                                 const float* __restrict__ tag_table,
                                 const float* __restrict__ char_table,
                                 const float* __restrict__ conv_w,
                                 const float* __restrict__ conv_b,
                                 __bf16* __restrict__ featb) {
  int row = blockIdx.x;          // row = s*B + b
  int s = row >> 6, b = row & 63;
  __bf16* out = featb + (size_t)row * KPAD;
  int t = threadIdx.x;           // 256 threads

  int xid = X[b * SS + s];
  for (int i = t; i < EE; i += 256) out[i] = (__bf16)embed_table[xid * EE + i];
  int tg = tags[b * SS + s];
  for (int i = t; i < TE; i += 256) out[EE + i] = (__bf16)tag_table[tg * TE + i];
  for (int i = KIN + t; i < KPAD; i += 256) out[i] = (__bf16)0.0f;  // K padding

  __shared__ float cbuf[CC][WW];
  for (int task = t; task < CC * WW; task += 256) {
    int cc = task >> 4, w = task & 15;
    float acc = conv_b[cc];
    int p = s * WW + w;
#pragma unroll
    for (int k = 0; k < 3; ++k) {
      int pp = p + k - 1;
      if (pp >= 0 && pp < SS * WW) {
        int ch = chars[b * SS * WW + pp];
        const float* ct = char_table + ch * CE;
        const float* cw = conv_w + cc * (CE * 3) + k;
        float a2 = 0.0f;
        for (int ci = 0; ci < CE; ++ci) a2 += ct[ci] * cw[ci * 3];
        acc += a2;
      }
    }
    cbuf[cc][w] = acc;
  }
  __syncthreads();
  for (int cc = t; cc < CC; cc += 256) {
    float m = cbuf[cc][0];
#pragma unroll
    for (int w = 1; w < WW; ++w) m = fmaxf(m, cbuf[cc][w]);
    out[EE + TE + cc] = (__bf16)m;
  }
}

// ---------------- 2) one-time weight conversion to bf16 + bias fuse ----------
__global__ void prep_kernel(const float* wih_f, const float* whh_f,
                            const float* bih_f, const float* bhh_f,
                            const float* wih_b, const float* whh_b,
                            const float* bih_b, const float* bhh_b,
                            const float* fcw,
                            __bf16* wih16, __bf16* whh16, __bf16* fcw16, float* bias) {
  long i = (long)blockIdx.x * 256 + threadIdx.x;
  const long NWIH = 2L * G4 * KPAD;       // padded K
  const long NWHH = 2L * G4 * HID;
  const long NFC  = (long)TGT * (2 * HID);
  const long NB   = 2L * G4;
  if (i < NWIH) {
    int d = (int)(i / (G4 * KPAD));
    long r = i % (G4 * KPAD);
    int g = (int)(r / KPAD), k = (int)(r % KPAD);
    const float* w = d ? wih_b : wih_f;
    wih16[i] = (__bf16)((k < KIN) ? w[(size_t)g * KIN + k] : 0.0f);
  } else if (i < NWIH + NWHH) {
    long j = i - NWIH;
    int d = (int)(j / (G4 * HID));
    long r = j % (G4 * HID);
    const float* w = d ? whh_b : whh_f;
    whh16[j] = (__bf16)w[r];
  } else if (i < NWIH + NWHH + NFC) {
    long j = i - NWIH - NWHH;
    fcw16[j] = (__bf16)fcw[j];
  } else if (i < NWIH + NWHH + NFC + NB) {
    long j = i - NWIH - NWHH - NFC;
    int d = (int)(j / G4); int g = (int)(j % G4);
    bias[j] = d ? (bih_b[g] + bhh_b[g]) : (bih_f[g] + bhh_f[g]);
  }
}

// ------- 3) input-gate precompute GEMM: 4 N-tiles per wave (A reused 4x) -----
//          fully unrolled K loop -> straight-line WMMA stream, no acc phis
__global__ void gates_gemm_kernel(const __bf16* __restrict__ featb,
                                  const __bf16* __restrict__ wih16,
                                  const float* __restrict__ bias,
                                  float* __restrict__ gpre) {
  int wave = threadIdx.x >> 5, lane = threadIdx.x & 31;
  int r = lane & 15, hi = lane >> 4;
  int idx = blockIdx.x * 8 + wave;        // 65536 wave-tiles
  int dir = idx >> 15;
  int rem = idx & 32767;
  int mt = rem >> 4;                      // 0..2047 row tiles over S*B
  int ntg = rem & 15;                     // group of 4 n-tiles
  int m0 = mt * 16;
  const __bf16* wbase = wih16 + (size_t)dir * G4 * KPAD;

  v8f acc[4];
#pragma unroll
  for (int t = 0; t < 4; ++t) {
    float bval = bias[dir * G4 + (ntg * 4 + t) * 16 + r];
#pragma unroll
    for (int j = 0; j < 8; ++j) acc[t][j] = bval;
  }

#pragma unroll
  for (int kc = 0; kc < KPAD / 32; ++kc) {
    int k0 = kc * 32;
    v16bf a = load_a_frag(featb, KPAD, m0, k0, r, hi);
#pragma unroll
    for (int t = 0; t < 4; ++t) {
      v16bf bf = load_b_fragT(wbase, KPAD, (ntg * 4 + t) * 16, k0, r, hi);
      acc[t] = __builtin_amdgcn_wmma_f32_16x16x32_bf16(false, a, false, bf,
                                                       (short)0, acc[t], false, false);
    }
  }
#pragma unroll
  for (int t = 0; t < 4; ++t) {
    float* out = gpre + gpre_base(dir, mt, ntg * 4 + t) + lane * 8;
#pragma unroll
    for (int j = 0; j < 8; ++j) out[j] = acc[t][j];
  }
}

// ---------------- 4) persistent recurrent LSTM (1 workgroup per direction) ---
__global__ __launch_bounds__(1024) void lstm_kernel(const float* __restrict__ gpre,
                                                    const __bf16* __restrict__ whh16,
                                                    float* __restrict__ Hout) {
  int dir = blockIdx.x;
  int tid = threadIdx.x;
  int wave = tid >> 5, lane = tid & 31, r = lane & 15, hi = lane >> 4;

  __shared__ __bf16 hbuf[BB * HPAD];
  for (int i = tid; i < BB * HPAD; i += 1024) hbuf[i] = (__bf16)0.0f;
  __syncthreads();

  const __bf16* wb = whh16 + (size_t)dir * G4 * HID;

  float c[2][8];
#pragma unroll
  for (int i = 0; i < 2; ++i)
#pragma unroll
    for (int j = 0; j < 8; ++j) c[i][j] = 0.0f;

  for (int s = 0; s < SS; ++s) {
    int seq = dir ? (SS - 1 - s) : s;

    // prefetch next step's gate tiles into cache (overlaps WMMA + activations)
    if (s + 1 < SS) {
      int seq2 = dir ? (SS - 2 - s) : (s + 1);
#pragma unroll
      for (int i = 0; i < 2; ++i) {
        int cidx = wave * 2 + i;
        int mt = cidx & 3, jt = cidx >> 2;
#pragma unroll
        for (int g = 0; g < 4; ++g) {
          const float* pf = gpre + gpre_base(dir, seq2 * 4 + mt, g * 16 + jt) + lane * 8;
          __builtin_prefetch(pf, 0, 2);
        }
      }
    }

    v8f acc[2][4];
#pragma unroll
    for (int i = 0; i < 2; ++i) {
      int cidx = wave * 2 + i;            // static combo: 4 batch-tiles x 16 j-tiles
      int mt = cidx & 3, jt = cidx >> 2;
      int m0 = mt * 16;
      // C init from precomputed gates: contiguous 32B per lane per gate tile
#pragma unroll
      for (int g = 0; g < 4; ++g) {
        const float* gin = gpre + gpre_base(dir, seq * 4 + mt, g * 16 + jt) + lane * 8;
#pragma unroll
        for (int j = 0; j < 8; ++j) acc[i][g][j] = gin[j];
      }
      // h @ W_hh^T, K=256 fully unrolled in 8 chunks, A shared across 4 gates
#pragma unroll
      for (int kc = 0; kc < HID / 32; ++kc) {
        int k0 = kc * 32;
        v16bf a;
        const __bf16* hp = hbuf + (m0 + r) * HPAD + k0 + hi * 8;
#pragma unroll
        for (int e = 0; e < 16; ++e) {
          int pr = e >> 1;
          int k = ((pr < 4) ? (pr * 2) : (16 + (pr - 4) * 2)) + (e & 1);
          a[e] = hp[k];
        }
#pragma unroll
        for (int g = 0; g < 4; ++g) {
          int n0 = g * HID + jt * 16;
          v16bf bfr;
          const __bf16* wp = wb + (size_t)(n0 + r) * HID + k0 + hi * 8;
#pragma unroll
          for (int e = 0; e < 16; ++e) bfr[e] = wp[(e < 8) ? e : (e + 8)];
          acc[i][g] = __builtin_amdgcn_wmma_f32_16x16x32_bf16(false, a, false, bfr,
                                                              (short)0, acc[i][g],
                                                              false, false);
        }
      }
    }
    __syncthreads();   // all reads of hbuf done
#pragma unroll
    for (int i = 0; i < 2; ++i) {
      int cidx = wave * 2 + i;
      int mt = cidx & 3, jt = cidx >> 2;
#pragma unroll
      for (int j = 0; j < 8; ++j) {
        int brow = mt * 16 + j + hi * 8;    // batch
        int col  = jt * 16 + r;             // hidden index
        float iv = sigf(acc[i][0][j]);
        float fv = sigf(acc[i][1][j]);
        float gv = tanhf(acc[i][2][j]);
        float ov = sigf(acc[i][3][j]);
        float cn = fv * c[i][j] + iv * gv;
        c[i][j] = cn;
        float hv = ov * tanhf(cn);
        hbuf[brow * HPAD + col] = (__bf16)hv;
        Hout[((size_t)seq * BB + brow) * (2 * HID) + dir * HID + col] = hv;
      }
    }
    __syncthreads();   // hbuf updated for next step
  }
}

// ---------------- 5) projection: P[S*B][16] = (H + lm*wd) @ fc_w^T + fc_b ----
__global__ void proj_kernel(const float* __restrict__ Hout,
                            const float* __restrict__ labels_m,
                            const float* __restrict__ wd,
                            const __bf16* __restrict__ fcw16,
                            const float* __restrict__ fc_b,
                            float* __restrict__ P) {
  int wave = threadIdx.x >> 5, lane = threadIdx.x & 31;
  int r = lane & 15, hi = lane >> 4;
  int tile = blockIdx.x * 8 + wave;  // 2048 tiles of 16 rows
  int m0 = tile * 16;
  int row = m0 + r;
  int s = row >> 6, b = row & 63;
  float lm = labels_m[b * SS + s];

  v8f acc;
  float bv = fc_b[r];
#pragma unroll
  for (int j = 0; j < 8; ++j) acc[j] = bv;

#pragma unroll
  for (int kc = 0; kc < (2 * HID) / 32; ++kc) {
    int k0 = kc * 32;
    v16bf a;
    const float* hp = Hout + (size_t)row * (2 * HID) + k0 + hi * 8;
    const float* wp = wd + k0 + hi * 8;
#pragma unroll
    for (int e = 0; e < 16; ++e) {
      int pr = e >> 1;
      int k = ((pr < 4) ? (pr * 2) : (16 + (pr - 4) * 2)) + (e & 1);
      a[e] = (__bf16)(hp[k] + lm * wp[k]);
    }
    v16bf bfr = load_b_fragT(fcw16, 2 * HID, 0, k0, r, hi);
    acc = __builtin_amdgcn_wmma_f32_16x16x32_bf16(false, a, false, bfr,
                                                  (short)0, acc, false, false);
  }
  float* out = P + (size_t)m0 * TGT;
#pragma unroll
  for (int j = 0; j < 8; ++j) out[(size_t)(j + hi * 8) * TGT + r] = acc[j];
}

// ---------------- 6) Viterbi forward scan (thread = (batch, cur-tag)) -------
__global__ __launch_bounds__(1024) void viterbi_fwd_kernel(const float* __restrict__ P,
                                                           const float* __restrict__ A,
                                                           float* __restrict__ deltas,
                                                           unsigned char* __restrict__ psis) {
  int cur = threadIdx.x & 15;
  int b = threadIdx.x >> 4;
  __shared__ float Ash[TGT * TGT];
  __shared__ float delta[BB * TGT];
  if (threadIdx.x < TGT * TGT) Ash[threadIdx.x] = A[threadIdx.x];
  float d0 = P[(size_t)b * TGT + cur];
  delta[b * TGT + cur] = d0;
  deltas[(size_t)b * TGT + cur] = d0;
  __syncthreads();
  for (int s = 1; s < SS; ++s) {
    float best = -3.0e38f; int barg = 0;
#pragma unroll
    for (int prev = 0; prev < TGT; ++prev) {
      float v = delta[b * TGT + prev] + Ash[prev * TGT + cur];
      if (v > best) { best = v; barg = prev; }
    }
    float nd = best + P[((size_t)s * BB + b) * TGT + cur];
    __syncthreads();
    delta[b * TGT + cur] = nd;
    deltas[((size_t)s * BB + b) * TGT + cur] = nd;
    psis[((size_t)s * BB + b) * TGT + cur] = (unsigned char)barg;
    __syncthreads();
  }
}

// ---------------- 7) backtrack: path (as float) + score -> d_out ------------
__global__ void viterbi_back_kernel(const float* __restrict__ deltas,
                                    const unsigned char* __restrict__ psis,
                                    float* __restrict__ out) {
  int b = threadIdx.x;
  if (b >= BB) return;
  const float* dL = deltas + ((size_t)(SS - 1) * BB + b) * TGT;
  int tag = 0; float best = dL[0];
#pragma unroll
  for (int t = 1; t < TGT; ++t) if (dL[t] > best) { best = dL[t]; tag = t; }
  float score = best;
  out[b * SS + (SS - 1)] = (float)tag;
  for (int s = SS - 1; s >= 1; --s) {
    int prev = psis[((size_t)s * BB + b) * TGT + tag];
    score += deltas[((size_t)(s - 1) * BB + b) * TGT + prev];
    out[b * SS + (s - 1)] = (float)prev;
    tag = prev;
  }
  out[BB * SS + b] = score;
}

// ---------------- host launcher ---------------------------------------------
extern "C" void kernel_launch(void* const* d_in, const int* in_sizes, int n_in,
                              void* d_out, int out_size, void* d_ws, size_t ws_size,
                              hipStream_t stream) {
  const int*   X        = (const int*)d_in[0];
  const float* labels_m = (const float*)d_in[1];
  const int*   tags     = (const int*)d_in[2];
  const int*   chars    = (const int*)d_in[3];
  const float* embed_t  = (const float*)d_in[4];
  const float* tag_t    = (const float*)d_in[5];
  const float* char_t   = (const float*)d_in[6];
  const float* conv_w   = (const float*)d_in[7];
  const float* conv_b   = (const float*)d_in[8];
  const float* w_ih_f   = (const float*)d_in[9];
  const float* w_hh_f   = (const float*)d_in[10];
  const float* b_ih_f   = (const float*)d_in[11];
  const float* b_hh_f   = (const float*)d_in[12];
  const float* w_ih_b   = (const float*)d_in[13];
  const float* w_hh_b   = (const float*)d_in[14];
  const float* b_ih_b   = (const float*)d_in[15];
  const float* b_hh_b   = (const float*)d_in[16];
  const float* fc_w     = (const float*)d_in[17];
  const float* fc_b     = (const float*)d_in[18];
  const float* Amat     = (const float*)d_in[19];
  const float* wd       = (const float*)d_in[20];

  char* ws = (char*)d_ws;
  size_t off = 0;
  __bf16* featb = (__bf16*)(ws + off); off += (size_t)SS * BB * KPAD * 2;        // 27.3 MB
  __bf16* wih16 = (__bf16*)(ws + off); off += (size_t)2 * G4 * KPAD * 2;         // 1.7 MB
  __bf16* whh16 = (__bf16*)(ws + off); off += (size_t)2 * G4 * HID * 2;          // 1.05 MB
  __bf16* fcw16 = (__bf16*)(ws + off); off += (size_t)TGT * 2 * HID * 2;         // 16 KB
  float*  bias  = (float*)(ws + off);  off += (size_t)2 * G4 * 4;                // 8 KB
  float*  gpre  = (float*)(ws + off);  off += (size_t)2 * SS * BB * G4 * 4;      // 268 MB
  float*  Hout  = (float*)(ws + off);  off += (size_t)SS * BB * 2 * HID * 4;     // 67 MB
  float*  P     = (float*)(ws + off);  off += (size_t)SS * BB * TGT * 4;         // 2.1 MB
  float*  delt  = (float*)(ws + off);  off += (size_t)SS * BB * TGT * 4;         // 2.1 MB
  unsigned char* psis = (unsigned char*)(ws + off);                              // 0.5 MB

  featurize_kernel<<<SS * BB, 256, 0, stream>>>(X, tags, chars, embed_t, tag_t,
                                                char_t, conv_w, conv_b, featb);

  long prep_n = 2L * G4 * KPAD + 2L * G4 * HID + (long)TGT * 2 * HID + 2L * G4;
  prep_kernel<<<(unsigned)((prep_n + 255) / 256), 256, 0, stream>>>(
      w_ih_f, w_hh_f, b_ih_f, b_hh_f, w_ih_b, w_hh_b, b_ih_b, b_hh_b,
      fc_w, wih16, whh16, fcw16, bias);

  gates_gemm_kernel<<<8192, 256, 0, stream>>>(featb, wih16, bias, gpre);

  lstm_kernel<<<2, 1024, 0, stream>>>(gpre, whh16, Hout);

  proj_kernel<<<256, 256, 0, stream>>>(Hout, labels_m, wd, fcw16, fc_b, P);

  viterbi_fwd_kernel<<<1, 1024, 0, stream>>>(P, Amat, delt, psis);

  viterbi_back_kernel<<<1, 64, 0, stream>>>(delt, psis, (float*)d_out);
}
